// CausalSelfAttention_4440996184532
// MI455X (gfx1250) — matrix operand order repaired and checked by reading, not compiled, not run on hardware
//
#include <hip/hip_runtime.h>
#include <hip/hip_bf16.h>

typedef __attribute__((ext_vector_type(16))) _Float16 v16h;
typedef __attribute__((ext_vector_type(8)))  _Float16 v8h;
typedef __attribute__((ext_vector_type(8)))  float    v8f;

#define NHEAD  16
#define HDIM   64
#define TSEQ   2048
#define DMODEL 1024

#define BK  32
#define LDT 40   // padded LDS stride in halves; 40*2=80B, multiple of 16B for b128 loads
#define LDK 72   // padded stride for 64-wide K/V tiles; 144B, multiple of 16B

// Join two 16-byte half-vectors into one WMMA operand register block.
__device__ __forceinline__ v16h frag_join(const _Float16* lo, const _Float16* hi) {
  union { v16h v; v8h h[2]; } u;
  u.h[0] = *(const v8h*)lo;
  u.h[1] = *(const v8h*)hi;
  return u.v;
}

// A fragment (16x32 f16) from row-major [M][stride] buffer; base = &buf[row0*stride + k0].
// Layout used here (contiguous k, not the ISA table order): lane m=l%16, g=l/16; halves i -> k = g*16 + i  (one contiguous 32B run).
__device__ __forceinline__ v16h load_afrag(const _Float16* base, int stride, int lane) {
  const int m = lane & 15, g = lane >> 4;
  const _Float16* p = base + (size_t)m * stride + g * 16;
  return frag_join(p, p + 8);
}

// B fragment (32x16 f16) from operand stored as [N][stride] (one row per output column).
// Layout used here (contiguous k, not the ISA table order): lane n=l%16, g=l/16; halves i -> k = g*16 + i  (one contiguous 32B run).
__device__ __forceinline__ v16h load_bfrag(const _Float16* base, int stride, int lane) {
  const int n = lane & 15, g = lane >> 4;
  const _Float16* p = base + (size_t)n * stride + g * 16;
  return frag_join(p, p + 8);
}

__device__ __forceinline__ v8f wmma16(v16h a, v16h b, v8f c) {
  return __builtin_amdgcn_wmma_f32_16x16x32_f16(false, a, false, b, (short)0, c, false, false);
}

// CDNA5 async global->LDS copy (16B per lane), tracked by ASYNCcnt.
// lds_off = low 32 bits of the generic shared-memory address.
__device__ __forceinline__ void async_copy_b128(unsigned lds_off, const _Float16* gptr) {
  asm volatile("global_load_async_to_lds_b128 %0, %1, off"
               :: "v"(lds_off), "v"(gptr) : "memory");
}
__device__ __forceinline__ void wait_asynccnt0() {
  asm volatile("s_wait_asynccnt 0x0" ::: "memory");
}

// ---------------------------------------------------------------------------
// Kernel 1: QKV projection. C[8192,3072] = x[8192,1024] @ w[1024,3072] + b.
// Scatter: q,k -> [B,H,T,64] f16 (row-major per key/query), v -> [B,H,64,T] f16 (transposed).
// ---------------------------------------------------------------------------
__global__ __launch_bounds__(256) void qkv_gemm_kernel(
    const float* __restrict__ x, const float* __restrict__ w,
    const float* __restrict__ bias,
    _Float16* __restrict__ qb, _Float16* __restrict__ kb, _Float16* __restrict__ vT)
{
  __shared__ _Float16 As[128 * LDT];
  __shared__ _Float16 Bs[128 * LDT];
  const int tid = threadIdx.x, lane = tid & 31, wave = tid >> 5;
  const int mblk = blockIdx.y * 128, nblk = blockIdx.x * 128;
  const int wm = (wave & 3) * 32, wn = (wave >> 2) * 64;
  const int K = DMODEL, N = 3 * DMODEL;

  v8f acc[2][4] = {};

  for (int k0 = 0; k0 < K; k0 += BK) {
    // Stage A tile 128x32 (f32 -> f16)
    {
      const int r = tid >> 3, c = (tid & 7) * 4;
#pragma unroll
      for (int p = 0; p < 4; ++p) {
        const int row = r + p * 32;
        float4 v = *(const float4*)(x + (size_t)(mblk + row) * K + k0 + c);
        _Float16* d = &As[row * LDT + c];
        d[0] = (_Float16)v.x; d[1] = (_Float16)v.y;
        d[2] = (_Float16)v.z; d[3] = (_Float16)v.w;
      }
    }
    // Stage B tile transposed: Bs[n][k] = w[k0+k][nblk+n] (f32 -> f16)
    {
      const int kk = tid >> 5, n4 = (tid & 31) * 4;
#pragma unroll
      for (int p = 0; p < 4; ++p) {
        const int kq = kk + p * 8;
        float4 v = *(const float4*)(w + (size_t)(k0 + kq) * N + nblk + n4);
        Bs[(n4 + 0) * LDT + kq] = (_Float16)v.x;
        Bs[(n4 + 1) * LDT + kq] = (_Float16)v.y;
        Bs[(n4 + 2) * LDT + kq] = (_Float16)v.z;
        Bs[(n4 + 3) * LDT + kq] = (_Float16)v.w;
      }
    }
    if (k0 + BK < K) {
      __builtin_prefetch(x + (size_t)(mblk + (tid >> 3)) * K + k0 + BK, 0, 0);
      __builtin_prefetch(w + (size_t)(k0 + BK + (tid >> 5)) * N + nblk, 0, 0);
    }
    __syncthreads();
    v16h af[2], bf[4];
#pragma unroll
    for (int tm = 0; tm < 2; ++tm) af[tm] = load_afrag(&As[(wm + tm * 16) * LDT], LDT, lane);
#pragma unroll
    for (int tn = 0; tn < 4; ++tn) bf[tn] = load_bfrag(&Bs[(wn + tn * 16) * LDT], LDT, lane);
#pragma unroll
    for (int tm = 0; tm < 2; ++tm)
#pragma unroll
      for (int tn = 0; tn < 4; ++tn) acc[tm][tn] = wmma16(af[tm], bf[tn], acc[tm][tn]);
    __syncthreads();
  }

  // Epilogue: bias + scatter into q / k / v^T f16 buffers.
  const int n16 = lane & 15, g = lane >> 4;
#pragma unroll
  for (int tm = 0; tm < 2; ++tm)
#pragma unroll
    for (int tn = 0; tn < 4; ++tn) {
      const int colg = nblk + wn + tn * 16 + n16;
      const int s = colg >> 10;          // 0=q, 1=k, 2=v (uniform per tile)
      const int rem = colg & 1023;
      const int h = rem >> 6, dd = rem & 63;
      const float bv = bias[colg];
#pragma unroll
      for (int r = 0; r < 8; ++r) {
        const int rowg = mblk + wm + tm * 16 + r + 8 * g;  // C layout: m = r + 8g
        const int b = rowg >> 11, t = rowg & 2047;
        const _Float16 val = (_Float16)(acc[tm][tn][r] + bv);
        const size_t bh = (size_t)(b * NHEAD + h);
        if (s == 0)      qb[(bh * TSEQ + t) * HDIM + dd] = val;
        else if (s == 1) kb[(bh * TSEQ + t) * HDIM + dd] = val;
        else             vT[(bh * HDIM + dd) * TSEQ + t] = val;
      }
    }
}

// ---------------------------------------------------------------------------
// Kernel 2: flash attention. Grid (T/128, B*H); 8 waves, 16 q-rows each.
// K/V tiles staged once per workgroup via async global->LDS copies (ASYNCcnt),
// uniform causal trip count; masked tail blocks contribute exactly zero.
// ---------------------------------------------------------------------------
__global__ __launch_bounds__(256) void attn_kernel(
    const _Float16* __restrict__ q, const _Float16* __restrict__ kmat,
    const _Float16* __restrict__ vT, _Float16* __restrict__ ao)
{
  __shared__ _Float16 Ks[64 * LDK];     // [key][dim]   (B-operand rows for Q K^T)
  __shared__ _Float16 Vs[64 * LDK];     // [dim][key]   (B-operand rows for P V)
  __shared__ _Float16 Ps[8][16 * 72];   // per-wave P staging, 144B row stride
  const int tid = threadIdx.x, lane = tid & 31, wave = tid >> 5;
  const int n16 = lane & 15, g = lane >> 4;
  const int bh = blockIdx.y, b = bh >> 4, h = bh & 15;
  const int qbase = blockIdx.x * 128 + wave * 16;

  const _Float16* qp = q + ((size_t)bh * TSEQ + qbase) * HDIM;
  v16h qf[2];
#pragma unroll
  for (int kb = 0; kb < 2; ++kb) qf[kb] = load_afrag(qp + kb * 32, HDIM, lane);

  v8f o[4] = {};
  float mrow[8], lrow[8];
#pragma unroll
  for (int r = 0; r < 8; ++r) { mrow[r] = -3.0e38f; lrow[r] = 0.0f; }
  const float scale = 0.125f;  // 1/sqrt(64)

  _Float16* pw = &Ps[wave][0];
  const unsigned ks_base = (unsigned)(uintptr_t)Ks;
  const unsigned vs_base = (unsigned)(uintptr_t)Vs;
  const int nkb = 2 * blockIdx.x + 2;   // keys [0, qblock_end) -- uniform across waves
  for (int blk = 0; blk < nkb; ++blk) {
    const int kbase = blk * 64;
    // Stage K tile (64 keys x 64 dims) and V^T tile (64 dims x 64 keys) into LDS.
    __syncthreads();   // all waves done reading previous tiles
#pragma unroll
    for (int p = 0; p < 2; ++p) {
      const int chunk = tid + p * 256;          // 512 chunks of 16B per tile
      const int row = chunk >> 3, c = (chunk & 7) * 8;
      async_copy_b128(ks_base + (unsigned)(row * LDK + c) * 2,
                      kmat + ((size_t)bh * TSEQ + kbase + row) * HDIM + c);
      async_copy_b128(vs_base + (unsigned)(row * LDK + c) * 2,
                      vT + ((size_t)bh * HDIM + row) * TSEQ + kbase + c);
    }
    wait_asynccnt0();
    __syncthreads();   // tiles visible to all waves

    // S = (Q K^T): 4 n-tiles x 2 k-chunks of WMMA, B-frags from LDS
    v8f S[4];
#pragma unroll
    for (int j = 0; j < 4; ++j) {
      v8f s = {};
      const _Float16* kp = &Ks[(j * 16 + n16) * LDK];
#pragma unroll
      for (int kb = 0; kb < 2; ++kb) {
        const _Float16* p = kp + kb * 32 + g * 16;
        s = wmma16(qf[kb], frag_join(p, p + 8), s);
      }
      S[j] = s;
    }
    // scale + causal mask (element (m=r+8g, n=n16) per C layout)
#pragma unroll
    for (int j = 0; j < 4; ++j) {
      const int kg = kbase + j * 16 + n16;
#pragma unroll
      for (int r = 0; r < 8; ++r) {
        const int qg = qbase + r + 8 * g;
        const float sv = S[j][r] * scale;
        S[j][r] = (kg <= qg) ? sv : -1.0e30f;
      }
    }
    // Online softmax: row stats reduced across the 16 lanes holding the row.
    float mnew[8], rsum[8];
#pragma unroll
    for (int r = 0; r < 8; ++r) {
      float mx = S[0][r];
#pragma unroll
      for (int j = 1; j < 4; ++j) mx = fmaxf(mx, S[j][r]);
#pragma unroll
      for (int off = 8; off >= 1; off >>= 1) mx = fmaxf(mx, __shfl_xor(mx, off, 32));
      mnew[r] = fmaxf(mrow[r], mx);
    }
#pragma unroll
    for (int r = 0; r < 8; ++r) {
      const float alpha = __expf(mrow[r] - mnew[r]);
      mrow[r] = mnew[r];
      lrow[r] *= alpha;
#pragma unroll
      for (int j = 0; j < 4; ++j) o[j][r] *= alpha;
      rsum[r] = 0.0f;
    }
#pragma unroll
    for (int j = 0; j < 4; ++j)
#pragma unroll
      for (int r = 0; r < 8; ++r) {
        const float p = __expf(S[j][r] - mrow[r]);
        S[j][r] = p;
        rsum[r] += p;
      }
#pragma unroll
    for (int r = 0; r < 8; ++r) {
#pragma unroll
      for (int off = 8; off >= 1; off >>= 1) rsum[r] += __shfl_xor(rsum[r], off, 32);
      lrow[r] += rsum[r];
    }
    // C-layout -> A-layout via per-wave LDS (DS ops in-order within wave: no barrier)
#pragma unroll
    for (int j = 0; j < 4; ++j)
#pragma unroll
      for (int r = 0; r < 8; ++r)
        pw[(r + 8 * g) * 72 + j * 16 + n16] = (_Float16)S[j][r];
    // O += P @ V   (Vs rows are d, columns are key index within the tile)
#pragma unroll
    for (int kb = 0; kb < 2; ++kb) {
      const v16h pa = load_afrag(pw + kb * 32, 72, lane);
#pragma unroll
      for (int j = 0; j < 4; ++j) {
        const _Float16* vp = &Vs[(j * 16 + n16) * LDK] + kb * 32 + g * 16;
        o[j] = wmma16(pa, frag_join(vp, vp + 8), o[j]);
      }
    }
  }
  // Normalize, store attn-out as [B,T,D] f16 (A operand of the proj GEMM)
#pragma unroll
  for (int r = 0; r < 8; ++r) {
    const float inv = 1.0f / lrow[r];
    const size_t row = (size_t)b * TSEQ + qbase + r + 8 * g;
#pragma unroll
    for (int j = 0; j < 4; ++j)
      ao[row * DMODEL + h * HDIM + j * 16 + n16] = (_Float16)(o[j][r] * inv);
  }
}

// ---------------------------------------------------------------------------
// Kernel 3: output projection. out[8192,1024] = ao[8192,1024] @ w_proj + b (f32 out).
// A tile is already f16 -> staged with async global->LDS copies.
// ---------------------------------------------------------------------------
__global__ __launch_bounds__(256) void proj_gemm_kernel(
    const _Float16* __restrict__ a, const float* __restrict__ w,
    const float* __restrict__ bias, float* __restrict__ out)
{
  __shared__ _Float16 As[128 * LDT];
  __shared__ _Float16 Bs[128 * LDT];
  const int tid = threadIdx.x, lane = tid & 31, wave = tid >> 5;
  const int mblk = blockIdx.y * 128, nblk = blockIdx.x * 128;
  const int wm = (wave & 3) * 32, wn = (wave >> 2) * 64;
  const int K = DMODEL, N = DMODEL;
  const unsigned as_base = (unsigned)(uintptr_t)As;

  v8f acc[2][4] = {};

  for (int k0 = 0; k0 < K; k0 += BK) {
    // Stage A tile 128x32 halves via async copies (512 x 16B chunks)
#pragma unroll
    for (int p = 0; p < 2; ++p) {
      const int chunk = tid + p * 256;
      const int row = chunk >> 2, c = (chunk & 3) * 8;
      async_copy_b128(as_base + (unsigned)(row * LDT + c) * 2,
                      a + (size_t)(mblk + row) * K + k0 + c);
    }
    // Stage B tile transposed (f32 -> f16)
    {
      const int kk = tid >> 5, n4 = (tid & 31) * 4;
#pragma unroll
      for (int p = 0; p < 4; ++p) {
        const int kq = kk + p * 8;
        float4 v = *(const float4*)(w + (size_t)(k0 + kq) * N + nblk + n4);
        Bs[(n4 + 0) * LDT + kq] = (_Float16)v.x;
        Bs[(n4 + 1) * LDT + kq] = (_Float16)v.y;
        Bs[(n4 + 2) * LDT + kq] = (_Float16)v.z;
        Bs[(n4 + 3) * LDT + kq] = (_Float16)v.w;
      }
    }
    if (k0 + BK < K)
      __builtin_prefetch(w + (size_t)(k0 + BK + (tid >> 5)) * N + nblk, 0, 0);
    wait_asynccnt0();
    __syncthreads();
    v16h af[2], bf[4];
#pragma unroll
    for (int tm = 0; tm < 2; ++tm) af[tm] = load_afrag(&As[(wm + tm * 16) * LDT], LDT, lane);
#pragma unroll
    for (int tn = 0; tn < 4; ++tn) bf[tn] = load_bfrag(&Bs[(wn + tn * 16) * LDT], LDT, lane);
#pragma unroll
    for (int tm = 0; tm < 2; ++tm)
#pragma unroll
      for (int tn = 0; tn < 4; ++tn) acc[tm][tn] = wmma16(af[tm], bf[tn], acc[tm][tn]);
    __syncthreads();
  }

  const int n16 = lane & 15, g = lane >> 4;
#pragma unroll
  for (int tm = 0; tm < 2; ++tm)
#pragma unroll
    for (int tn = 0; tn < 4; ++tn) {
      const int colg = nblk + wn + tn * 16 + n16;
      const float bv = bias[colg];
#pragma unroll
      for (int r = 0; r < 8; ++r) {
        const int rowg = mblk + wm + tm * 16 + r + 8 * g;
        out[(size_t)rowg * N + colg] = acc[tm][tn][r] + bv;
      }
    }
}

// ---------------------------------------------------------------------------
extern "C" void kernel_launch(void* const* d_in, const int* in_sizes, int n_in,
                              void* d_out, int out_size, void* d_ws, size_t ws_size,
                              hipStream_t stream) {
  const float* x      = (const float*)d_in[0];
  const float* w_qkv  = (const float*)d_in[1];
  const float* b_qkv  = (const float*)d_in[2];
  const float* w_proj = (const float*)d_in[3];
  const float* b_proj = (const float*)d_in[4];
  float* out = (float*)d_out;

  char* ws = (char*)d_ws;
  const size_t SZ = (size_t)4 * NHEAD * TSEQ * HDIM * sizeof(_Float16);  // 16 MB each
  _Float16* qbuf = (_Float16*)(ws);
  _Float16* kbuf = (_Float16*)(ws + SZ);
  _Float16* vT   = (_Float16*)(ws + 2 * SZ);
  _Float16* ao   = (_Float16*)(ws + 3 * SZ);

  dim3 blk(256);
  // QKV: N=3072 -> 24 col blocks, M=8192 -> 64 row blocks
  qkv_gemm_kernel<<<dim3(24, 64), blk, 0, stream>>>(x, w_qkv, b_qkv, qbuf, kbuf, vT);
  // Attention: (T/128, B*H)
  attn_kernel<<<dim3(TSEQ / 128, 4 * NHEAD), blk, 0, stream>>>(qbuf, kbuf, vT, ao);
  // Projection: N=1024 -> 8 col blocks
  proj_gemm_kernel<<<dim3(8, 64), blk, 0, stream>>>(ao, w_proj, b_proj, out);
}
